// MakeMoE_66073776881830
// MI455X (gfx1250) — matrix operand order repaired and checked by reading, not compiled, not run on hardware
//
#include <hip/hip_runtime.h>
#include <stdint.h>
#include <stddef.h>

#define B_TOK  4096
#define DMODEL 1024
#define DFF    4096
#define NEXP   8

typedef __attribute__((ext_vector_type(16))) __bf16 v16bf;
typedef __attribute__((ext_vector_type(8)))  __bf16 v8bf;
typedef __attribute__((ext_vector_type(8)))  float  v8f;
typedef __attribute__((ext_vector_type(4)))  float  f4;
typedef __attribute__((ext_vector_type(2)))  unsigned u2;
typedef __attribute__((address_space(3)))    __bf16 lds_bf16;

// two fp32 -> packed 2 x bf16 (native cvt, RTNE)
__device__ __forceinline__ unsigned pack2(float lo, float hi) {
    unsigned short a = __builtin_bit_cast(unsigned short, (__bf16)lo);
    unsigned short b = __builtin_bit_cast(unsigned short, (__bf16)hi);
    return (unsigned)a | ((unsigned)b << 16);
}

__device__ __forceinline__ v16bf cat16(v8bf lo, v8bf hi) {
    return __builtin_shufflevector(lo, hi, 0, 1, 2, 3, 4, 5, 6, 7,
                                           8, 9, 10, 11, 12, 13, 14, 15);
}

__global__ void moe_zero_counts(int* __restrict__ counts) {
    if (threadIdx.x < NEXP) counts[threadIdx.x] = 0;
}

__global__ void moe_route(const int* __restrict__ idx,
                          int* __restrict__ counts,
                          int* __restrict__ token_of) {
    int b = blockIdx.x * blockDim.x + threadIdx.x;
    if (b >= B_TOK) return;
    int e = idx[b];
    int slot = atomicAdd(&counts[e], 1);
    token_of[e * B_TOK + slot] = b;
}

// -------- Layer 1: h[token] = relu(x[token] @ W1[e] + b1[e]),  bf16 out --------
__launch_bounds__(128)
__global__ void moe_ffn1(const float* __restrict__ x,
                         const float* __restrict__ W1,
                         const float* __restrict__ b1,
                         const int* __restrict__ counts,
                         const int* __restrict__ token_of,
                         __bf16* __restrict__ h) {
    const int e   = blockIdx.z;
    const int cnt = counts[e];
    const int m0  = blockIdx.y * 64;
    if (m0 >= cnt) return;                 // uniform: WMMA region stays full-EXEC
    const int n0  = blockIdx.x * 64;

    __shared__ __bf16 As[64][40];          // [m][k], stride 40 (80B, 16B-aligned rows)
    __shared__ __bf16 Bs[64][40];          // [n][k], transposed for vector frag loads
    __shared__ int    tok[64];

    const int tid   = threadIdx.x;
    const int lane  = tid & 31;
    const int wave  = tid >> 5;
    const int waveM = (wave >> 1) * 32;
    const int waveN = (wave & 1) * 32;

    if (tid < 64) {
        int s = m0 + tid;
        tok[tid] = (s < cnt) ? token_of[e * B_TOK + s] : -1;
    }
    __syncthreads();

    // A-tile loader: float4 per row, 4 rows/thread
    const int c4  = (tid & 7) * 4;         // k within tile
    const int rb4 = tid >> 3;              // row base 0..15
    int toff[4];                           // clamped token row offsets (elements)
    #pragma unroll
    for (int i = 0; i < 4; ++i) {
        int t = tok[rb4 + i * 16];
        toff[i] = (t < 0 ? 0 : t) * DMODEL;   // clamp: safe read, rows never stored
    }
    // B-tile loader: float4 along n, 4 k-rows/thread
    const int nq  = (tid & 15) * 4;        // n within tile
    const int kb8 = tid >> 4;              // k base 0..7

    v8f acc[2][2] = {};
    const float* W1e = W1 + (size_t)e * DMODEL * DFF;

    for (int k0 = 0; k0 < DMODEL; k0 += 32) {
        // A tile 64x32 (gathered), fp32 -> packed bf16x4 (b64 LDS store)
        #pragma unroll
        for (int i = 0; i < 4; ++i) {
            f4 v = *(const f4*)(x + toff[i] + k0 + c4);
            u2 p;
            p.x = pack2(v.x, v.y);
            p.y = pack2(v.z, v.w);
            *(u2*)&As[rb4 + i * 16][c4] = p;
        }
        // B tile 32(k)x64(n) -> transposed LDS [n][k]
        #pragma unroll
        for (int i = 0; i < 4; ++i) {
            int kr = kb8 + i * 8;
            f4 v = *(const f4*)(W1e + (size_t)(k0 + kr) * DFF + n0 + nq);
            Bs[nq][kr]     = (__bf16)v.x;
            Bs[nq + 1][kr] = (__bf16)v.y;
            Bs[nq + 2][kr] = (__bf16)v.z;
            Bs[nq + 3][kr] = (__bf16)v.w;
        }
        if (k0 + 32 < DMODEL) {            // prefetch next K-slab
            __builtin_prefetch(W1e + (size_t)(k0 + 32 + kb8) * DFF + n0 + nq, 0, 1);
            __builtin_prefetch(x + toff[0] + k0 + 32 + c4, 0, 1);
        }
        __syncthreads();

        // Fragments per ISA 16-bit layouts (CDNA5 ISA §7.12.2), vector LDS loads
        const int l15 = lane & 15;
        const int hiL = lane >> 4;
        v16bf a[2], b[2];
        #pragma unroll
        for (int mt = 0; mt < 2; ++mt) {
            const __bf16* ap = &As[waveM + mt * 16 + l15][0];
            a[mt] = cat16(*(const v8bf*)(ap + hiL * 8),
                          *(const v8bf*)(ap + 16 + hiL * 8));
        }
        #pragma unroll
        for (int nt = 0; nt < 2; ++nt) {
            const __bf16* bp = &Bs[waveN + nt * 16 + l15][0];
            b[nt] = cat16(*(const v8bf*)(bp + hiL * 8), *(const v8bf*)(bp + 16 + hiL * 8));
        }
        #pragma unroll
        for (int mt = 0; mt < 2; ++mt)
            #pragma unroll
            for (int nt = 0; nt < 2; ++nt)
                acc[mt][nt] = __builtin_amdgcn_wmma_f32_16x16x32_bf16(
                    false, a[mt], false, b[nt], (short)0, acc[mt][nt], false, false);
        __syncthreads();
    }

    // Epilogue: +bias, ReLU, bf16 scatter to h[token]
    const int hiL = lane >> 4;
    const int nl  = lane & 15;
    #pragma unroll
    for (int mt = 0; mt < 2; ++mt) {
        #pragma unroll
        for (int nt = 0; nt < 2; ++nt) {
            int gcol = n0 + waveN + nt * 16 + nl;
            float bias = b1[e * DFF + gcol];
            #pragma unroll
            for (int r = 0; r < 8; ++r) {
                int rowin = waveM + mt * 16 + hiL * 8 + r;
                int t = tok[rowin];
                if (t >= 0) {
                    float v = acc[mt][nt][r] + bias;
                    v = v > 0.0f ? v : 0.0f;
                    h[(size_t)t * DFF + gcol] = (__bf16)v;
                }
            }
        }
    }
}

// -------- Layer 2: out[token] = h[token] @ W2[e] + b2[e],  fp32 out --------
__launch_bounds__(128)
__global__ void moe_ffn2(const __bf16* __restrict__ h,
                         const float* __restrict__ W2,
                         const float* __restrict__ b2,
                         const int* __restrict__ counts,
                         const int* __restrict__ token_of,
                         float* __restrict__ out) {
    const int e   = blockIdx.z;
    const int cnt = counts[e];
    const int m0  = blockIdx.y * 64;
    if (m0 >= cnt) return;
    const int n0  = blockIdx.x * 64;

    __shared__ __bf16 As[64][40];
    __shared__ __bf16 Bs[64][40];          // [n][k] transposed
    __shared__ int    tok[64];

    const int tid   = threadIdx.x;
    const int lane  = tid & 31;
    const int wave  = tid >> 5;
    const int waveM = (wave >> 1) * 32;
    const int waveN = (wave & 1) * 32;

    if (tid < 64) {
        int s = m0 + tid;
        tok[tid] = (s < cnt) ? token_of[e * B_TOK + s] : -1;
    }
    __syncthreads();

    // A-tile async loader: 8 bf16 (16 B) per issue, 2 rows/thread
    const int c8  = (tid & 3) * 8;         // k within tile
    const int rb2 = tid >> 2;              // row base 0..31
    int toff[2];
    #pragma unroll
    for (int i = 0; i < 2; ++i) {
        int t = tok[rb2 + i * 32];
        toff[i] = (t < 0 ? 0 : t) * DFF;
    }
    const int nq  = (tid & 15) * 4;
    const int kb8 = tid >> 4;

    v8f acc[2][2] = {};
    const float* W2e = W2 + (size_t)e * DFF * DMODEL;

    for (int k0 = 0; k0 < DFF; k0 += 32) {
        // A tile 64x32 from bf16 h: async global->LDS b128 copies (ASYNCcnt)
        #pragma unroll
        for (int i = 0; i < 2; ++i) {
            const __bf16* gp = h + toff[i] + k0 + c8;
            unsigned loff = (unsigned)(size_t)(lds_bf16*)&As[rb2 + i * 32][c8];
            unsigned long long ga = (unsigned long long)(size_t)gp;
            asm volatile("global_load_async_to_lds_b128 %0, %1, off"
                         :: "v"(loff), "v"(ga) : "memory");
        }
        // B tile 32(k)x64(n) -> transposed LDS [n][k]
        #pragma unroll
        for (int i = 0; i < 4; ++i) {
            int kr = kb8 + i * 8;
            f4 v = *(const f4*)(W2e + (size_t)(k0 + kr) * DMODEL + n0 + nq);
            Bs[nq][kr]     = (__bf16)v.x;
            Bs[nq + 1][kr] = (__bf16)v.y;
            Bs[nq + 2][kr] = (__bf16)v.z;
            Bs[nq + 3][kr] = (__bf16)v.w;
        }
        if (k0 + 32 < DFF) {
            __builtin_prefetch(W2e + (size_t)(k0 + 32 + kb8) * DMODEL + n0 + nq, 0, 1);
            __builtin_prefetch(h + toff[0] + k0 + 32 + c8, 0, 1);
        }
        asm volatile("s_wait_asynccnt 0" ::: "memory");
        __syncthreads();

        const int l15 = lane & 15;
        const int hiL = lane >> 4;
        v16bf a[2], b[2];
        #pragma unroll
        for (int mt = 0; mt < 2; ++mt) {
            const __bf16* ap = &As[waveM + mt * 16 + l15][0];
            a[mt] = cat16(*(const v8bf*)(ap + hiL * 8),
                          *(const v8bf*)(ap + 16 + hiL * 8));
        }
        #pragma unroll
        for (int nt = 0; nt < 2; ++nt) {
            const __bf16* bp = &Bs[waveN + nt * 16 + l15][0];
            b[nt] = cat16(*(const v8bf*)(bp + hiL * 8), *(const v8bf*)(bp + 16 + hiL * 8));
        }
        #pragma unroll
        for (int mt = 0; mt < 2; ++mt)
            #pragma unroll
            for (int nt = 0; nt < 2; ++nt)
                acc[mt][nt] = __builtin_amdgcn_wmma_f32_16x16x32_bf16(
                    false, a[mt], false, b[nt], (short)0, acc[mt][nt], false, false);
        __syncthreads();
    }

    const int hiL = lane >> 4;
    const int nl  = lane & 15;
    #pragma unroll
    for (int mt = 0; mt < 2; ++mt) {
        #pragma unroll
        for (int nt = 0; nt < 2; ++nt) {
            int gcol = n0 + waveN + nt * 16 + nl;
            float bias = b2[e * DMODEL + gcol];
            #pragma unroll
            for (int r = 0; r < 8; ++r) {
                int rowin = waveM + mt * 16 + hiL * 8 + r;
                int t = tok[rowin];
                if (t >= 0)
                    out[(size_t)t * DMODEL + gcol] = acc[mt][nt][r] + bias;
            }
        }
    }
}

extern "C" void kernel_launch(void* const* d_in, const int* in_sizes, int n_in,
                              void* d_out, int out_size, void* d_ws, size_t ws_size,
                              hipStream_t stream) {
    const float* x  = (const float*)d_in[0];
    const float* W1 = (const float*)d_in[1];
    const float* b1 = (const float*)d_in[2];
    const float* W2 = (const float*)d_in[3];
    const float* b2 = (const float*)d_in[4];
    const int*   mi = (const int*)d_in[5];

    char* ws = (char*)d_ws;
    int*    counts   = (int*)ws;                                  // 8 ints
    int*    token_of = (int*)(ws + 256);                          // E*B ints (128 KB)
    __bf16* hbuf     = (__bf16*)(ws + 256 + (size_t)NEXP * B_TOK * sizeof(int)); // 32 MB

    moe_zero_counts<<<1, 32, 0, stream>>>(counts);
    moe_route<<<B_TOK / 256, 256, 0, stream>>>(mi, counts, token_of);

    dim3 g1(DFF / 64, B_TOK / 64, NEXP);     // 64 x 64 x 8
    moe_ffn1<<<g1, 128, 0, stream>>>(x, W1, b1, counts, token_of, hbuf);

    dim3 g2(DMODEL / 64, B_TOK / 64, NEXP);  // 16 x 64 x 8
    moe_ffn2<<<g2, 128, 0, stream>>>(hbuf, W2, b2, counts, token_of, (float*)d_out);
}
